// ConvNet_2396591751238
// MI455X (gfx1250) — matrix operand order repaired and checked by reading, not compiled, not run on hardware
//
#include <hip/hip_runtime.h>
#include <hip/hip_bf16.h>
#include <math.h>

// ---------------- problem constants (match reference) ----------------
#define B_GR   256
#define M_G    256
#define NN     (B_GR * M_G)   // 65536 nodes
#define KNN    15
#define F_INP  7
#define C_CH   128
#define C2     768
#define NL     3
#define LEAKY  0.01f
#define BN_EPS 1e-5f

typedef __attribute__((ext_vector_type(16))) _Float16 v16h;
typedef __attribute__((ext_vector_type(8)))  _Float16 v8h;
typedef __attribute__((ext_vector_type(8)))  float    v8f;

#define GAS __attribute__((address_space(1)))

__device__ inline v8h ldg_v8h(const _Float16* p) {
    return *(const GAS v8h*)p;
}
__device__ inline v16h ldg_v16h(const _Float16* p) {
    return *(const GAS v16h*)p;
}
__device__ inline void stg_v8h(_Float16* p, v8h v) {
    *(GAS v8h*)p = v;
}
__device__ inline void stg_h(_Float16* p, _Float16 v) {
    *(GAS _Float16*)p = v;
}

// ---------------------------------------------------------------------
// Pad x [N,7] f32 -> xp [N,32] f16 (cols 7..31 zeroed) so layer-1 GEMM
// runs one unguarded K=32 WMMA step.
// ---------------------------------------------------------------------
__global__ void xpad_kernel(const float* __restrict__ x, _Float16* __restrict__ xp) {
    int i = blockIdx.x * blockDim.x + threadIdx.x;
    if (i >= NN) return;
#pragma unroll
    for (int c = 0; c < 32; ++c)
        stg_h(xp + (size_t)i * 32 + c,
              (c < F_INP) ? (_Float16)x[(size_t)i * F_INP + c] : (_Float16)0.f);
}

// ---------------------------------------------------------------------
// Pack f32 weights [K x ldw] (cols 0..Ncols-1) into the
// per-lane-contiguous WMMA B-fragment order, K zero-padded to Kp:
//   Wp[((kt*CT + ct)*32 + lane)*16 + e] = W[kt*32 + (lane>>4)*8 + (e&7) + (e>>3)*16][ct*16 + (lane&15)]
// GEMM then fetches one contiguous v16h (32B) per lane per k-step.
// ---------------------------------------------------------------------
__global__ void pack_w_kernel(const float* __restrict__ W, int K, int Kp, int ldw,
                              int Ncols, _Float16* __restrict__ Wp) {
    const int CT = Ncols >> 4;
    int t = blockIdx.x * blockDim.x + threadIdx.x;
    int total = Kp * Ncols;            // == (Kp/32)*CT*512 halves
    if (t >= total) return;
    int tile = t >> 9;
    int r    = t & 511;
    int lane = r >> 4, e = r & 15;
    int kt = tile / CT, ct = tile % CT;
    int k = kt * 32 + ((lane >> 4) << 3) + (e & 7) + ((e >> 3) << 4);
    int c = ct * 16 + (lane & 15);
    float v = (k < K) ? W[(size_t)k * ldw + c] : 0.f;
    stg_h(Wp + t, (_Float16)v);
}

// ---------------------------------------------------------------------
// Per-graph kNN on columns 0..3. One block per graph, one thread per node.
// Positions staged in LDS; register top-15 via worst-slot replacement.
// knn stride padded to 16 ints per node.
// ---------------------------------------------------------------------
__global__ __launch_bounds__(M_G)
void knn_kernel(const float* __restrict__ x, int* __restrict__ knn) {
    __shared__ float4 pos[M_G];
    const int b = blockIdx.x;
    const int m = threadIdx.x;
    const float* xr = x + (size_t)(b * M_G + m) * F_INP;
    float4 p = make_float4(xr[0], xr[1], xr[2], xr[3]);
    pos[m] = p;
    __syncthreads();

    float bd[KNN];
    int   bi[KNN];
#pragma unroll
    for (int k = 0; k < KNN; ++k) { bd[k] = 3.0e38f; bi[k] = 0; }
    float worst = 3.0e38f; int ws = 0;

    for (int j = 0; j < M_G; ++j) {
        if (j == m) continue;
        float4 q = pos[j];
        float dx = p.x - q.x, dy = p.y - q.y, dz = p.z - q.z, dw = p.w - q.w;
        float d2 = dx * dx + dy * dy + dz * dz + dw * dw;
        if (d2 < worst) {
            bd[ws] = d2; bi[ws] = j;
            worst = -1.0f;
#pragma unroll
            for (int k = 0; k < KNN; ++k)
                if (bd[k] > worst) { worst = bd[k]; ws = k; }
        }
    }
    int* o = knn + (size_t)(b * M_G + m) * 16;
#pragma unroll
    for (int k = 0; k < KNN; ++k) o[k] = b * M_G + bi[k];
    o[15] = 0;
}

// ---------------------------------------------------------------------
// Neighbor-average: hout[i] = (1/15) * sum_k hin[knn[i,k]].
// (Every dst has exactly 15 in-edges => gcn norm == 1/15.)
// One thread handles 8 consecutive f16 channels (16B global vector loads).
// ---------------------------------------------------------------------
template <int LDA>
__global__ void knn_agg_kernel(const _Float16* __restrict__ hin,
                               const int* __restrict__ knn,
                               _Float16* __restrict__ hout) {
    const int groups = LDA / 8;
    int gid  = blockIdx.x * blockDim.x + threadIdx.x;
    int node = gid / groups;
    int g    = gid % groups;
    if (node >= NN) return;

    const int* nb = knn + (size_t)node * 16;
    float acc[8];
#pragma unroll
    for (int c = 0; c < 8; ++c) acc[c] = 0.f;

    for (int k = 0; k < KNN; ++k) {
        int j = nb[k];
        v8h v = ldg_v8h(hin + (size_t)j * LDA + g * 8);
#pragma unroll
        for (int c = 0; c < 8; ++c) acc[c] += (float)v[c];
    }
    v8h out;
#pragma unroll
    for (int c = 0; c < 8; ++c) out[c] = (_Float16)(acc[c] * (1.0f / 15.0f));
    stg_v8h(hout + (size_t)node * LDA + g * 8, out);
}

// ---------------------------------------------------------------------
// A-fragment (16x32 f16, ISA 7.12.2): lane%16 = row, and the lane's K
// indices are two contiguous 8-half runs: k0+hi*8+0..7 and k0+16+hi*8+0..7.
// => two global_load_b128 per fragment, no guards (lda multiple of 32).
// ---------------------------------------------------------------------
__device__ inline v16h load_a_frag(const _Float16* __restrict__ Arow /* +m*lda+hi*8 */,
                                   int k0) {
    v8h lo = ldg_v8h(Arow + k0);
    v8h hi = ldg_v8h(Arow + k0 + 16);
    return __builtin_shufflevector(lo, hi, 0, 1, 2, 3, 4, 5, 6, 7,
                                           8, 9, 10, 11, 12, 13, 14, 15);
}

// ---------------------------------------------------------------------
// Fused TAGConv layer: out = lrelu( h0@W[0] + h1@W[1] + h2@W[2] + bias )
// Wp packed per-hop (hop stride Kp*128 halves), CT = 8 col tiles.
// grid.x = NN/16 row tiles; 8 waves/block cover 128 cols.
// ---------------------------------------------------------------------
__global__ __launch_bounds__(256)
void tag_gemm_kernel(const _Float16* __restrict__ h0,
                     const _Float16* __restrict__ h1,
                     const _Float16* __restrict__ h2,
                     int Kp, int lda,
                     const _Float16* __restrict__ Wp,
                     const float* __restrict__ bias,
                     _Float16* __restrict__ out) {
    const int lane = threadIdx.x & 31;
    const int wave = threadIdx.x >> 5;       // = ct (col tile), CT = 8
    const int row0 = blockIdx.x * 16;
    const int col0 = wave * 16;
    const int n = lane & 15, hi = lane >> 4;
    const int KT = Kp >> 5;

    v8f acc = {};
    const _Float16* hs[3] = { h0, h1, h2 };
    for (int t = 0; t < 3; ++t) {
        const _Float16* Arow = hs[t] + (size_t)(row0 + (lane & 15)) * lda + hi * 8;
        const _Float16* Wt = Wp + (size_t)t * Kp * C_CH
                               + ((size_t)wave * 32 + lane) * 16;
        for (int kt = 0; kt < KT; ++kt) {
            v16h a  = load_a_frag(Arow, kt * 32);
            v16h bf = ldg_v16h(Wt + (size_t)kt * 8 * 512);
            acc = __builtin_amdgcn_wmma_f32_16x16x32_f16(
                false, a, false, bf, (short)0, acc, false, false);
        }
    }
#pragma unroll
    for (int r = 0; r < 8; ++r) {
        int row = row0 + (hi << 3) + r;
        float v = acc[r] + bias[col0 + n];
        v = (v > 0.f) ? v : LEAKY * v;
        stg_h(out + (size_t)row * C_CH + col0 + n, (_Float16)v);
    }
}

// ---------------------------------------------------------------------
// MLP layer: out = lrelu( A[256,768] @ W[768,768] + b ), f16 in/out.
// Wp packed with CT = 48 col tiles. grid = (16, 6), 8 waves/block.
// ---------------------------------------------------------------------
__global__ __launch_bounds__(256)
void mlp_gemm_kernel(const _Float16* __restrict__ A,
                     const _Float16* __restrict__ Wp,
                     const float* __restrict__ bias,
                     _Float16* __restrict__ out) {
    const int lane = threadIdx.x & 31;
    const int wave = threadIdx.x >> 5;
    const int row0 = blockIdx.x * 16;
    const int ct   = blockIdx.y * 8 + wave;  // 0..47
    const int col0 = ct * 16;
    const int n = lane & 15, hi = lane >> 4;
    const int CT = C2 / 16;                  // 48
    const int KT = C2 / 32;                  // 24

    const _Float16* Arow = A + (size_t)(row0 + n) * C2 + hi * 8;
    const _Float16* Wt   = Wp + ((size_t)ct * 32 + lane) * 16;

    v8f acc = {};
    for (int kt = 0; kt < KT; ++kt) {
        v16h a  = load_a_frag(Arow, kt * 32);
        v16h bf = ldg_v16h(Wt + (size_t)kt * CT * 512);
        acc = __builtin_amdgcn_wmma_f32_16x16x32_f16(
            false, a, false, bf, (short)0, acc, false, false);
    }
#pragma unroll
    for (int r = 0; r < 8; ++r) {
        int row = row0 + (hi << 3) + r;
        float v = acc[r] + bias[col0 + n];
        v = (v > 0.f) ? v : LEAKY * v;
        stg_h(out + (size_t)row * C2 + col0 + n, (_Float16)v);
    }
}

// ---------------------------------------------------------------------
// Per-graph mean+max pool of h [NN,128] into g [256,768] at layer offset.
// ---------------------------------------------------------------------
__global__ __launch_bounds__(C_CH)
void pool_kernel(const _Float16* __restrict__ h, float* __restrict__ g, int layerOff) {
    const int b = blockIdx.x, c = threadIdx.x;
    const _Float16* base = h + (size_t)b * M_G * C_CH + c;
    float s = 0.f, mx = -3.0e38f;
    for (int m = 0; m < M_G; ++m) {
        float v = (float)*(const GAS _Float16*)(base + (size_t)m * C_CH);
        s += v;
        mx = fmaxf(mx, v);
    }
    g[(size_t)b * C2 + layerOff + c]        = s * (1.0f / (float)M_G);
    g[(size_t)b * C2 + layerOff + C_CH + c] = mx;
}

// ---------------------------------------------------------------------
// BatchNorm over the batch axis (256 rows) per column; writes f16.
// ---------------------------------------------------------------------
__global__ __launch_bounds__(B_GR)
void bn_kernel(const float* __restrict__ g, const float* __restrict__ gamma,
               const float* __restrict__ beta, _Float16* __restrict__ gh) {
    __shared__ float red[B_GR];
    const int col = blockIdx.x, t = threadIdx.x;
    float v = g[(size_t)t * C2 + col];
    red[t] = v;
    __syncthreads();
    for (int s = B_GR / 2; s > 0; s >>= 1) {
        if (t < s) red[t] += red[t + s];
        __syncthreads();
    }
    float mu = red[0] * (1.0f / (float)B_GR);
    __syncthreads();
    float d = v - mu;
    red[t] = d * d;
    __syncthreads();
    for (int s = B_GR / 2; s > 0; s >>= 1) {
        if (t < s) red[t] += red[t + s];
        __syncthreads();
    }
    float var = red[0] * (1.0f / (float)B_GR);
    float o = d * rsqrtf(var + BN_EPS) * gamma[col] + beta[col];
    stg_h(gh + (size_t)t * C2 + col, (_Float16)o);
}

// ---------------------------------------------------------------------
// Head: y = gh @ Wo + bo ; tanh on first two columns. 768 outputs.
// ---------------------------------------------------------------------
__global__ void head_kernel(const _Float16* __restrict__ gh,
                            const float* __restrict__ Wo,
                            const float* __restrict__ bo,
                            float* __restrict__ y) {
    int idx = blockIdx.x * blockDim.x + threadIdx.x;
    if (idx >= B_GR * NL) return;
    int b = idx / NL, o = idx % NL;
    float s = bo[o];
    for (int k0 = 0; k0 < C2; k0 += 8) {
        v8h v = ldg_v8h(gh + (size_t)b * C2 + k0);
#pragma unroll
        for (int c = 0; c < 8; ++c)
            s += (float)v[c] * Wo[(size_t)(k0 + c) * NL + o];
    }
    if (o < 2) s = tanhf(s);
    y[idx] = s;
}

// ---------------------------------------------------------------------
extern "C" void kernel_launch(void* const* d_in, const int* in_sizes, int n_in,
                              void* d_out, int out_size, void* d_ws, size_t ws_size,
                              hipStream_t stream) {
    (void)in_sizes; (void)n_in; (void)out_size; (void)ws_size;

    const float* x    = (const float*)d_in[0];
    const float* Wc1  = (const float*)d_in[1];
    const float* bc1  = (const float*)d_in[2];
    const float* Wc2  = (const float*)d_in[3];
    const float* bc2  = (const float*)d_in[4];
    const float* Wc3  = (const float*)d_in[5];
    const float* bc3  = (const float*)d_in[6];
    const float* bn_g = (const float*)d_in[7];
    const float* bn_b = (const float*)d_in[8];
    const float* Wm[5] = { (const float*)d_in[9],  (const float*)d_in[11],
                           (const float*)d_in[13], (const float*)d_in[15],
                           (const float*)d_in[17] };
    const float* bm[5] = { (const float*)d_in[10], (const float*)d_in[12],
                           (const float*)d_in[14], (const float*)d_in[16],
                           (const float*)d_in[18] };
    const float* Wo   = (const float*)d_in[19];
    const float* bo   = (const float*)d_in[20];
    float* y = (float*)d_out;

    // ---- workspace carve-up (~95 MB) ----
    char* w = (char*)d_ws;
    int*      knn = (int*)w;       w += (size_t)NN * 16 * sizeof(int);
    _Float16* xp  = (_Float16*)w;  w += (size_t)NN * 32 * sizeof(_Float16);
    _Float16* x1  = (_Float16*)w;  w += (size_t)NN * 32 * sizeof(_Float16);
    _Float16* x2  = (_Float16*)w;  w += (size_t)NN * 32 * sizeof(_Float16);
    _Float16* hA  = (_Float16*)w;  w += (size_t)NN * C_CH * sizeof(_Float16);
    _Float16* hB  = (_Float16*)w;  w += (size_t)NN * C_CH * sizeof(_Float16);
    _Float16* hC  = (_Float16*)w;  w += (size_t)NN * C_CH * sizeof(_Float16);
    _Float16* hD  = (_Float16*)w;  w += (size_t)NN * C_CH * sizeof(_Float16);
    float*    g   = (float*)w;     w += (size_t)B_GR * C2 * sizeof(float);
    _Float16* gh0 = (_Float16*)w;  w += (size_t)B_GR * C2 * sizeof(_Float16);
    _Float16* gh1 = (_Float16*)w;  w += (size_t)B_GR * C2 * sizeof(_Float16);
    _Float16* wp1 = (_Float16*)w;  w += (size_t)3 * 32 * C_CH * sizeof(_Float16);
    _Float16* wp2 = (_Float16*)w;  w += (size_t)3 * C_CH * C_CH * sizeof(_Float16);
    _Float16* wp3 = (_Float16*)w;  w += (size_t)3 * C_CH * C_CH * sizeof(_Float16);
    _Float16* wpm[5];
    for (int l = 0; l < 5; ++l) { wpm[l] = (_Float16*)w; w += (size_t)C2 * C2 * sizeof(_Float16); }

    // ---- weight packing (f32 -> f16, WMMA B-fragment order) ----
    for (int t = 0; t < 3; ++t) {
        pack_w_kernel<<<(32 * C_CH + 255) / 256, 256, 0, stream>>>(
            Wc1 + (size_t)t * F_INP * C_CH, F_INP, 32, C_CH, C_CH, wp1 + (size_t)t * 32 * C_CH);
        pack_w_kernel<<<(C_CH * C_CH + 255) / 256, 256, 0, stream>>>(
            Wc2 + (size_t)t * C_CH * C_CH, C_CH, C_CH, C_CH, C_CH, wp2 + (size_t)t * C_CH * C_CH);
        pack_w_kernel<<<(C_CH * C_CH + 255) / 256, 256, 0, stream>>>(
            Wc3 + (size_t)t * C_CH * C_CH, C_CH, C_CH, C_CH, C_CH, wp3 + (size_t)t * C_CH * C_CH);
    }
    for (int l = 0; l < 5; ++l)
        pack_w_kernel<<<(C2 * C2 + 255) / 256, 256, 0, stream>>>(
            Wm[l], C2, C2, C2, C2, wpm[l]);

    // ---- graph construction ----
    xpad_kernel<<<NN / 256, 256, 0, stream>>>(x, xp);
    knn_kernel<<<B_GR, M_G, 0, stream>>>(x, knn);

    // ---- TAGConv layer 1 (K=7 padded to 32) ----
    knn_agg_kernel<32><<<NN * 4 / 256, 256, 0, stream>>>(xp, knn, x1);
    knn_agg_kernel<32><<<NN * 4 / 256, 256, 0, stream>>>(x1, knn, x2);
    tag_gemm_kernel<<<NN / 16, 256, 0, stream>>>(xp, x1, x2, 32, 32, wp1, bc1, hA);
    pool_kernel<<<B_GR, C_CH, 0, stream>>>(hA, g, 0);

    // ---- TAGConv layer 2 (K=128) ----
    knn_agg_kernel<C_CH><<<NN * (C_CH / 8) / 256, 256, 0, stream>>>(hA, knn, hB);
    knn_agg_kernel<C_CH><<<NN * (C_CH / 8) / 256, 256, 0, stream>>>(hB, knn, hC);
    tag_gemm_kernel<<<NN / 16, 256, 0, stream>>>(hA, hB, hC, C_CH, C_CH, wp2, bc2, hD);
    pool_kernel<<<B_GR, C_CH, 0, stream>>>(hD, g, 2 * C_CH);

    // ---- TAGConv layer 3 ----
    knn_agg_kernel<C_CH><<<NN * (C_CH / 8) / 256, 256, 0, stream>>>(hD, knn, hA);
    knn_agg_kernel<C_CH><<<NN * (C_CH / 8) / 256, 256, 0, stream>>>(hA, knn, hB);
    tag_gemm_kernel<<<NN / 16, 256, 0, stream>>>(hD, hA, hB, C_CH, C_CH, wp3, bc3, hC);
    pool_kernel<<<B_GR, C_CH, 0, stream>>>(hC, g, 4 * C_CH);

    // ---- batchnorm -> f16 ----
    bn_kernel<<<C2, B_GR, 0, stream>>>(g, bn_g, bn_b, gh0);

    // ---- 5-layer MLP (ping-pong gh0/gh1) ----
    dim3 mgrid(B_GR / 16, C2 / (8 * 16));  // (16, 6)
    _Float16* cur = gh0;
    _Float16* nxt = gh1;
    for (int l = 0; l < 5; ++l) {
        mlp_gemm_kernel<<<mgrid, 256, 0, stream>>>(cur, wpm[l], bm[l], nxt);
        _Float16* t = cur; cur = nxt; nxt = t;
    }

    // ---- head (768 -> 3, tanh on first 2 cols) ----
    head_kernel<<<(B_GR * NL + 255) / 256, 256, 0, stream>>>(cur, Wo, bo, y);
}
